// Decoder_80238579024561
// MI455X (gfx1250) — matrix operand order repaired and checked by reading, not compiled, not run on hardware
//
#include <hip/hip_runtime.h>
#include <hip/hip_bf16.h>

#define VOCAB   32000
#define DIM_EMB 512
#define DIM_H   1024
#define DIM_Z   256
#define SL      64
#define BS      64
#define FOURH   (4 * DIM_H)   // 4096
#define MROWS   (SL * BS)     // 4096

typedef _Float16 half_t;
typedef __attribute__((ext_vector_type(16))) _Float16     v16h;
typedef __attribute__((ext_vector_type(8)))  float        v8f;
typedef __attribute__((ext_vector_type(4)))  unsigned int u32x4;

union FragU { u32x4 q[2]; v16h v; };

// A-matrix 16x32 f16 fragment (ISA 7.12.2): lane L holds row m0+(L&15);
// lanes 0-15: halves = K[kk+0..7], K[kk+16..23]; lanes 16-31: +8.
__device__ __forceinline__ v16h load_a_frag(const half_t* __restrict__ A, int lda,
                                            int row, int kk, int lane) {
    const int kb = kk + ((lane & 16) ? 8 : 0);
    const half_t* p = A + (size_t)row * lda + kb;
    FragU u;
    u.q[0] = *(const u32x4*)(p);        // K kb..kb+7
    u.q[1] = *(const u32x4*)(p + 16);   // K kb+16..kb+23
    return u.v;
}

// B-matrix 32x16 fragment built from row-major W[n][k] (we compute A @ W^T):
// lane L holds column n = col; lanes 0-15: K[kk+0..7], K[kk+16..23]; lanes 16-31: +8.
__device__ __forceinline__ v16h load_bt_frag(const half_t* __restrict__ W, int ldw,
                                             int col, int kk, int lane) {
    const int kb = kk + ((lane & 16) ? 8 : 0);
    const half_t* p = W + (size_t)col * ldw + kb;
    FragU u;
    u.q[0] = *(const u32x4*)(p);        // K kb..kb+7
    u.q[1] = *(const u32x4*)(p + 16);   // K kb+16..kb+23
    return u.v;
}

__device__ __forceinline__ v8f wmma16x16x32(v16h a, v16h b, v8f c) {
    // (neg_a, A, neg_b, B, c_mod, C, reuse_a, reuse_b)
    return __builtin_amdgcn_wmma_f32_16x16x32_f16(false, a, false, b, (short)0, c,
                                                  false, false);
}

// ---------------------------------------------------------------- small prep
__global__ __launch_bounds__(256) void cvt_f16_kernel(const float* __restrict__ in,
                                                      half_t* __restrict__ out, int n) {
    int i = blockIdx.x * 256 + threadIdx.x;
    if (i < n) out[i] = (half_t)in[i];
}

__global__ __launch_bounds__(256) void bias_sum_kernel(const float* __restrict__ a,
                                                       const float* __restrict__ b,
                                                       float* __restrict__ o, int n) {
    int i = blockIdx.x * 256 + threadIdx.x;
    if (i < n) o[i] = a[i] + b[i];
}

__global__ __launch_bounds__(256) void init_state_kernel(float* __restrict__ c,
                                                         half_t* __restrict__ h, int n) {
    int i = blockIdx.x * 256 + threadIdx.x;
    if (i < n) { c[i] = 0.0f; h[i] = (half_t)0.0f; }
}

// zemb[b][e] = cat(z_f,z_c)[b] . z2emb_W[e] + z2emb_b[e]
__global__ __launch_bounds__(256) void zemb_kernel(const float* __restrict__ z_c,
                                                   const float* __restrict__ z_f,
                                                   const float* __restrict__ W,
                                                   const float* __restrict__ bias,
                                                   float* __restrict__ zemb) {
    int idx = blockIdx.x * 256 + threadIdx.x;      // 64*512
    int b = idx >> 9, e = idx & 511;
    const float* wr = W + (size_t)e * DIM_Z;
    float s = bias[e];
    #pragma unroll 4
    for (int k = 0; k < 128; ++k) s += z_f[b * 128 + k] * wr[k];
    #pragma unroll 4
    for (int k = 0; k < 128; ++k) s += z_c[b * 128 + k] * wr[128 + k];
    zemb[idx] = s;
}

// x[t][b][e] = f16(embed_W[tok][e] + zemb[b][e]); tok = t==0 ? input_ids[0][b] : target[t-1][b]
__global__ __launch_bounds__(256) void build_x_kernel(const int* __restrict__ input_ids,
                                                      const int* __restrict__ target,
                                                      const float* __restrict__ embed_W,
                                                      const float* __restrict__ zemb,
                                                      half_t* __restrict__ x) {
    int idx = blockIdx.x * 256 + threadIdx.x;      // 64*64*512
    int e = idx & 511;
    int tb = idx >> 9;
    int b = tb & 63, t = tb >> 6;
    int tok = (t == 0) ? input_ids[b] : target[(t - 1) * BS + b];
    float v = embed_W[(size_t)tok * DIM_EMB + e] + zemb[b * DIM_EMB + e];
    x[idx] = (half_t)v;
}

// ------------------------------------------------- pre-gates: X @ W_ih^T + b
// block = 8 waves as 2(M) x 4(N); each wave owns a 32x32 tile (4 WMMA / k-step)
__global__ __launch_bounds__(256) void pre_gemm_kernel(const half_t* __restrict__ X,
                                                       const half_t* __restrict__ Wih,
                                                       const float* __restrict__ bsum,
                                                       float* __restrict__ pre) {
    const int lane = threadIdx.x & 31;
    const int wave = threadIdx.x >> 5;
    const int m0 = blockIdx.y * 64  + (wave >> 2) * 32;
    const int n0 = blockIdx.x * 128 + (wave & 3) * 32;
    const int ar = m0 + (lane & 15);
    const int bc = n0 + (lane & 15);
    v8f acc00 = {0.f,0.f,0.f,0.f,0.f,0.f,0.f,0.f};
    v8f acc01 = acc00, acc10 = acc00, acc11 = acc00;
    for (int kk = 0; kk < DIM_EMB; kk += 32) {
        v16h a0 = load_a_frag(X, DIM_EMB, ar,      kk, lane);
        v16h a1 = load_a_frag(X, DIM_EMB, ar + 16, kk, lane);
        v16h b0 = load_bt_frag(Wih, DIM_EMB, bc,      kk, lane);
        v16h b1 = load_bt_frag(Wih, DIM_EMB, bc + 16, kk, lane);
        acc00 = wmma16x16x32(a0, b0, acc00);
        acc01 = wmma16x16x32(a0, b1, acc01);
        acc10 = wmma16x16x32(a1, b0, acc10);
        acc11 = wmma16x16x32(a1, b1, acc11);
    }
    const int rb = (lane & 16) ? 8 : 0;
    const int c0 = n0 + (lane & 15);
    const float bb0 = bsum[c0], bb1 = bsum[c0 + 16];
    #pragma unroll
    for (int r = 0; r < 8; ++r) {
        const size_t row0 = (size_t)(m0 + rb + r);
        const size_t row1 = row0 + 16;
        pre[row0 * FOURH + c0]      = acc00[r] + bb0;
        pre[row0 * FOURH + c0 + 16] = acc01[r] + bb1;
        pre[row1 * FOURH + c0]      = acc10[r] + bb0;
        pre[row1 * FOURH + c0 + 16] = acc11[r] + bb1;
    }
}

// ------------------------------- per-step: gates = pre[t] + h @ W_hh^T (M=64)
__global__ __launch_bounds__(256) void step_gemm_kernel(const half_t* __restrict__ H,
                                                        const half_t* __restrict__ Whh,
                                                        const float* __restrict__ pre_t,
                                                        float* __restrict__ gates) {
    const int lane = threadIdx.x & 31;
    const int wave = threadIdx.x >> 5;
    const int m0 = (wave >> 1) * 16;                 // batch rows 0..63
    const int n0 = blockIdx.x * 64 + (wave & 1) * 32;
    const int arow = m0 + (lane & 15);
    const int bcol = n0 + (lane & 15);
    const int rbase = m0 + ((lane & 16) ? 8 : 0);
    const int c0 = n0 + (lane & 15);
    v8f acc0, acc1;                                  // C initialized from pre-gates
    #pragma unroll
    for (int r = 0; r < 8; ++r) {
        acc0[r] = pre_t[(size_t)(rbase + r) * FOURH + c0];
        acc1[r] = pre_t[(size_t)(rbase + r) * FOURH + c0 + 16];
    }
    for (int kk = 0; kk < DIM_H; kk += 32) {
        v16h a  = load_a_frag(H, DIM_H, arow, kk, lane);
        v16h b0 = load_bt_frag(Whh, DIM_H, bcol,      kk, lane);
        v16h b1 = load_bt_frag(Whh, DIM_H, bcol + 16, kk, lane);
        acc0 = wmma16x16x32(a, b0, acc0);
        acc1 = wmma16x16x32(a, b1, acc1);
    }
    #pragma unroll
    for (int r = 0; r < 8; ++r) {
        gates[(size_t)(rbase + r) * FOURH + c0]      = acc0[r];
        gates[(size_t)(rbase + r) * FOURH + c0 + 16] = acc1[r];
    }
}

__global__ __launch_bounds__(256) void lstm_cell_kernel(const float* __restrict__ gates,
                                                        float* __restrict__ c,
                                                        half_t* __restrict__ h16,
                                                        half_t* __restrict__ hs_t) {
    const int idx = blockIdx.x * 256 + threadIdx.x;  // 64*1024
    const int b = idx >> 10, j = idx & (DIM_H - 1);
    const size_t base = (size_t)b * FOURH + j;
    const float gi = gates[base];
    const float gf = gates[base + DIM_H];
    const float gg = gates[base + 2 * DIM_H];
    const float go = gates[base + 3 * DIM_H];
    const float si = 1.0f / (1.0f + __expf(-gi));
    const float sf = 1.0f / (1.0f + __expf(-gf));
    const float so = 1.0f / (1.0f + __expf(-go));
    const float cn = sf * c[idx] + si * tanhf(gg);
    c[idx] = cn;
    const float h = so * tanhf(cn);
    h16[idx]  = (half_t)h;
    hs_t[idx] = (half_t)h;
}

// ----------------------------------- projection: hs @ proj_W^T + b -> logits
// block = 8 waves as 2(M) x 4(N); 32x32 tile per wave; NT stores for the
// 524MB write-once output so proj weights stay resident in the 192MB L2.
__global__ __launch_bounds__(256) void proj_gemm_kernel(const half_t* __restrict__ HS,
                                                        const half_t* __restrict__ PW,
                                                        const float* __restrict__ pb,
                                                        float* __restrict__ out) {
    const int lane = threadIdx.x & 31;
    const int wave = threadIdx.x >> 5;
    const int m0 = blockIdx.y * 64  + (wave >> 2) * 32;
    const int n0 = blockIdx.x * 128 + (wave & 3) * 32;
    const int ar = m0 + (lane & 15);
    const int bc = n0 + (lane & 15);
    v8f acc00 = {0.f,0.f,0.f,0.f,0.f,0.f,0.f,0.f};
    v8f acc01 = acc00, acc10 = acc00, acc11 = acc00;
    for (int kk = 0; kk < DIM_H; kk += 32) {
        v16h a0 = load_a_frag(HS, DIM_H, ar,      kk, lane);
        v16h a1 = load_a_frag(HS, DIM_H, ar + 16, kk, lane);
        v16h b0 = load_bt_frag(PW, DIM_H, bc,      kk, lane);
        v16h b1 = load_bt_frag(PW, DIM_H, bc + 16, kk, lane);
        acc00 = wmma16x16x32(a0, b0, acc00);
        acc01 = wmma16x16x32(a0, b1, acc01);
        acc10 = wmma16x16x32(a1, b0, acc10);
        acc11 = wmma16x16x32(a1, b1, acc11);
    }
    const int rb = (lane & 16) ? 8 : 0;
    const int c0 = n0 + (lane & 15);
    const float bb0 = pb[c0], bb1 = pb[c0 + 16];
    #pragma unroll
    for (int r = 0; r < 8; ++r) {
        const size_t row0 = (size_t)(m0 + rb + r);
        const size_t row1 = row0 + 16;
        __builtin_nontemporal_store(acc00[r] + bb0, &out[row0 * VOCAB + c0]);
        __builtin_nontemporal_store(acc01[r] + bb1, &out[row0 * VOCAB + c0 + 16]);
        __builtin_nontemporal_store(acc10[r] + bb0, &out[row1 * VOCAB + c0]);
        __builtin_nontemporal_store(acc11[r] + bb1, &out[row1 * VOCAB + c0 + 16]);
    }
}

extern "C" void kernel_launch(void* const* d_in, const int* in_sizes, int n_in,
                              void* d_out, int out_size, void* d_ws, size_t ws_size,
                              hipStream_t stream) {
    (void)in_sizes; (void)n_in; (void)out_size; (void)ws_size;

    const float* z_c      = (const float*)d_in[0];
    const float* z_f      = (const float*)d_in[1];
    const int*   input_ids= (const int*)  d_in[2];
    const int*   target   = (const int*)  d_in[3];
    const float* embed_W  = (const float*)d_in[4];
    const float* z2emb_W  = (const float*)d_in[5];
    const float* z2emb_b  = (const float*)d_in[6];
    const float* W_ih     = (const float*)d_in[7];
    const float* W_hh     = (const float*)d_in[8];
    const float* b_ih     = (const float*)d_in[9];
    const float* b_hh     = (const float*)d_in[10];
    const float* proj_W   = (const float*)d_in[11];
    const float* proj_b   = (const float*)d_in[12];
    float* out = (float*)d_out;

    // workspace carve-up (all offsets 256B aligned)
    char* p = (char*)d_ws;
    auto alloc = [&](size_t bytes) {
        char* r = p; p += (bytes + 255) & ~(size_t)255; return r;
    };
    half_t* x_h    = (half_t*)alloc((size_t)MROWS * DIM_EMB * 2);       //  4 MB
    half_t* Wih_h  = (half_t*)alloc((size_t)FOURH * DIM_EMB * 2);       //  4 MB
    half_t* Whh_h  = (half_t*)alloc((size_t)FOURH * DIM_H * 2);        //  8 MB
    half_t* proj_h = (half_t*)alloc((size_t)VOCAB * DIM_H * 2);        // 64 MB
    half_t* hs_h   = (half_t*)alloc((size_t)MROWS * DIM_H * 2);        //  8 MB
    half_t* h_h    = (half_t*)alloc((size_t)BS * DIM_H * 2);
    float*  zemb   = (float*)alloc((size_t)BS * DIM_EMB * 4);
    float*  bsum   = (float*)alloc((size_t)FOURH * 4);
    float*  pre    = (float*)alloc((size_t)MROWS * FOURH * 4);         // 64 MB
    float*  gates  = (float*)alloc((size_t)BS * FOURH * 4);            //  1 MB
    float*  cstate = (float*)alloc((size_t)BS * DIM_H * 4);

    // weight conversions + bias sum
    cvt_f16_kernel<<<(FOURH * DIM_EMB + 255) / 256, 256, 0, stream>>>(W_ih, Wih_h, FOURH * DIM_EMB);
    cvt_f16_kernel<<<(FOURH * DIM_H  + 255) / 256, 256, 0, stream>>>(W_hh, Whh_h, FOURH * DIM_H);
    cvt_f16_kernel<<<(VOCAB * DIM_H  + 255) / 256, 256, 0, stream>>>(proj_W, proj_h, VOCAB * DIM_H);
    bias_sum_kernel<<<(FOURH + 255) / 256, 256, 0, stream>>>(b_ih, b_hh, bsum, FOURH);
    init_state_kernel<<<(BS * DIM_H + 255) / 256, 256, 0, stream>>>(cstate, h_h, BS * DIM_H);

    // inputs: zemb, then x = embed[tokens] + zemb  (f16)
    zemb_kernel<<<(BS * DIM_EMB + 255) / 256, 256, 0, stream>>>(z_c, z_f, z2emb_W, z2emb_b, zemb);
    build_x_kernel<<<(MROWS * DIM_EMB + 255) / 256, 256, 0, stream>>>(input_ids, target, embed_W,
                                                                      zemb, x_h);

    // hoisted input GEMM: pre = x @ W_ih^T + (b_ih + b_hh)   [4096 x 4096]
    pre_gemm_kernel<<<dim3(FOURH / 128, MROWS / 64), 256, 0, stream>>>(x_h, Wih_h, bsum, pre);

    // sequential recurrence: 64 x { gates = pre[t] + h @ W_hh^T ; LSTM cell }
    for (int t = 0; t < SL; ++t) {
        step_gemm_kernel<<<dim3(FOURH / 64, 1), 256, 0, stream>>>(
            h_h, Whh_h, pre + (size_t)t * BS * FOURH, gates);
        lstm_cell_kernel<<<(BS * DIM_H + 255) / 256, 256, 0, stream>>>(
            gates, cstate, h_h, hs_h + (size_t)t * BS * DIM_H);
    }

    // projection: logits = hs @ proj_W^T + proj_b  [4096 x 32000]
    proj_gemm_kernel<<<dim3(VOCAB / 128, MROWS / 64), 256, 0, stream>>>(hs_h, proj_h, proj_b, out);
}
